// fasterRCNNFPNBase_15899968930081
// MI455X (gfx1250) — matrix operand order repaired and checked by reading, not compiled, not run on hardware
//
#include <hip/hip_runtime.h>
#include <hip/hip_bf16.h>

// ---------------------------------------------------------------------------
// Types for WMMA fragments
// ---------------------------------------------------------------------------
typedef __attribute__((ext_vector_type(16))) _Float16 v16h;
typedef __attribute__((ext_vector_type(8)))  _Float16 v8h;
typedef __attribute__((ext_vector_type(8)))  float    v8f;
typedef __attribute__((ext_vector_type(4)))  float    v4f;

#define N_ROIS 512
#define CCH    256
#define PP     7
#define DFEAT  12544   // 256*7*7
#define HID    1024
#define NCLS   81
#define NREG   324
#define NLOG   405     // 81 + 324

static __device__ __forceinline__ v16h cat8(v8h lo, v8h hi) {
    v16h r;
#pragma unroll
    for (int i = 0; i < 8; ++i) { r[i] = lo[i]; r[i + 8] = hi[i]; }
    return r;
}

static __device__ __forceinline__ void wait_async0() {
#if __has_builtin(__builtin_amdgcn_s_wait_asynccnt)
    __builtin_amdgcn_s_wait_asynccnt(0);
#else
    asm volatile("s_wait_asynccnt 0" ::: "memory");
#endif
}

// Async global -> LDS copy of 16 bytes per lane (CDNA5, tracked by ASYNCcnt)
static __device__ __forceinline__ void async_b128(const void* gsrc, void* ldst) {
    unsigned lds_off = (unsigned)(unsigned long long)(uintptr_t)ldst;
    asm volatile("global_load_async_to_lds_b128 %0, %1, off"
                 :: "v"(lds_off), "v"(gsrc)
                 : "memory");
}

// ---------------------------------------------------------------------------
// Kernel 1: ROI-Align (legacy, aligned=false), writes f16 activation matrix
// grid = N_ROIS blocks, 256 threads (1 channel per thread)
// ---------------------------------------------------------------------------
__global__ void roi_align_kernel(const float* __restrict__ p2,
                                 const float* __restrict__ p3,
                                 const float* __restrict__ p4,
                                 const float* __restrict__ p5,
                                 const float* __restrict__ rois,
                                 _Float16* __restrict__ x0) {
    __shared__ int   s_ylo[14], s_yhi[14], s_xlo[14], s_xhi[14];
    __shared__ float s_yf[14], s_xf[14];
    __shared__ int   s_yv[14], s_xv[14];

    const int r = blockIdx.x;
    const int c = threadIdx.x;           // channel

    const float bidx = rois[r * 5 + 0];
    const float rx1  = rois[r * 5 + 1];
    const float ry1  = rois[r * 5 + 2];
    const float rx2  = rois[r * 5 + 3];
    const float ry2  = rois[r * 5 + 4];

    // FPN level selection (uses unscaled roi coords)
    const float rhv = ry2 - ry1 + 1.0f;
    const float rwv = rx2 - rx1 + 1.0f;
    int lvl = (int)floorf(log2f(sqrtf(rhv * rwv) / 56.0f + 1e-6f));
    lvl = lvl < 0 ? 0 : (lvl > 3 ? 3 : lvl);

    const float* feats[4] = { p2, p3, p4, p5 };
    const int H = 200 >> lvl;            // 200,100,50,25
    const int W = H;
    const float scale = 1.0f / (float)(4 << lvl);

    const float x1 = rx1 * scale, y1 = ry1 * scale;
    const float x2 = rx2 * scale, y2 = ry2 * scale;
    const float rw = fmaxf(x2 - x1, 1.0f);
    const float rh = fmaxf(y2 - y1, 1.0f);

    // sample coordinates (uniform per block) -> LDS
    if (threadIdx.x < 14) {
        const int s = threadIdx.x;
        const float g = (float)(s >> 1) + 0.25f + 0.5f * (float)(s & 1);
        const float y = y1 + g * (rh / 7.0f);
        const int v = (y >= -1.0f) && (y <= (float)H);
        const float cy = fminf(fmaxf(y, 0.0f), (float)(H - 1));
        const int lo = (int)floorf(cy);
        const int hi = (lo + 1 < H - 1) ? lo + 1 : H - 1;
        s_ylo[s] = lo; s_yhi[s] = hi; s_yf[s] = cy - (float)lo; s_yv[s] = v;
    } else if (threadIdx.x >= 32 && threadIdx.x < 46) {
        const int s = threadIdx.x - 32;
        const float g = (float)(s >> 1) + 0.25f + 0.5f * (float)(s & 1);
        const float x = x1 + g * (rw / 7.0f);
        const int v = (x >= -1.0f) && (x <= (float)W);
        const float cx = fminf(fmaxf(x, 0.0f), (float)(W - 1));
        const int lo = (int)floorf(cx);
        const int hi = (lo + 1 < W - 1) ? lo + 1 : W - 1;
        s_xlo[s] = lo; s_xhi[s] = hi; s_xf[s] = cx - (float)lo; s_xv[s] = v;
    }
    __syncthreads();

    const int b = (int)bidx;
    const float* __restrict__ f0 =
        feats[lvl] + (size_t)(b * CCH + c) * (size_t)(H * W);
    _Float16* __restrict__ xo = x0 + (size_t)r * DFEAT + c * (PP * PP);

#pragma unroll
    for (int py = 0; py < PP; ++py) {
#pragma unroll
        for (int px = 0; px < PP; ++px) {
            float acc = 0.0f;
#pragma unroll
            for (int a = 0; a < 2; ++a) {
                const int sy = 2 * py + a;
#pragma unroll
                for (int bb = 0; bb < 2; ++bb) {
                    const int sx = 2 * px + bb;
                    if (s_yv[sy] && s_xv[sx]) {
                        const int yl = s_ylo[sy], yh = s_yhi[sy];
                        const int xl = s_xlo[sx], xh = s_xhi[sx];
                        const float fy = s_yf[sy], fx = s_xf[sx];
                        const float v00 = f0[(size_t)yl * W + xl];
                        const float v01 = f0[(size_t)yl * W + xh];
                        const float v10 = f0[(size_t)yh * W + xl];
                        const float v11 = f0[(size_t)yh * W + xh];
                        acc += (1.0f - fy) * ((1.0f - fx) * v00 + fx * v01)
                             +         fy  * ((1.0f - fx) * v10 + fx * v11);
                    }
                }
            }
            xo[py * PP + px] = (_Float16)(acc * 0.25f);
        }
    }
}

// ---------------------------------------------------------------------------
// Generic NT GEMM:  C[M,N] = A[M,K](f16) * B[N,K](f32, converted to f16)^T
// 64x64 block tile, 8 waves (2 M x 4 N), wave does 2x(16x16) via WMMA f16.
// Double-buffered LDS: A tile via GLOBAL_LOAD_ASYNC_TO_LDS_B128 (ASYNCcnt),
// B tile via global load + v_cvt_pk_f16_f32 + ds_store (needs conversion).
// ---------------------------------------------------------------------------
#define LDSH 40   // 32 + 8 halves padding (80B rows, 16B aligned)

template <bool RELU_F16OUT>
__global__ __launch_bounds__(256)
void gemm_nt(const _Float16* __restrict__ A, const float* __restrict__ B,
             const float* __restrict__ bias, void* __restrict__ Cout,
             int N, int K, int lda, int ldb, int ldc) {
    __shared__ _Float16 lA[2][64 * LDSH];
    __shared__ _Float16 lB[2][64 * LDSH];

    const int nBase = blockIdx.x * 64;
    const int mBase = blockIdx.y * 64;

    const int wave = threadIdx.x >> 5;
    const int lane = threadIdx.x & 31;
    const int wm = wave >> 2;      // 0..1  (32 rows each)
    const int wn = wave & 3;       // 0..3  (16 cols each)
    const int l16 = lane & 15;
    const int hb  = lane >> 4;     // 0 or 1

    v8f acc0 = {};
    v8f acc1 = {};

    const int sRow = threadIdx.x >> 2;   // 0..63
    const int sCid = threadIdx.x & 3;    // 0..3

    const _Float16* aSrc = A + (size_t)(mBase + sRow) * lda + sCid * 8;
    const int gn = nBase + sRow;
    const bool bValid = gn < N;
    const float* bSrc = B + (size_t)(bValid ? gn : 0) * ldb + sCid * 8;
    const int ldsOff = sRow * LDSH + sCid * 8;

    // ---- staging helpers ----
    auto asyncA = [&](int buf, int k0) {
        async_b128(aSrc + k0, &lA[buf][ldsOff]);
    };
    auto loadB = [&](int k0) -> v8h {
        v4f f0 = *(const v4f*)(bSrc + k0);
        v4f f1 = *(const v4f*)(bSrc + k0 + 4);
        __builtin_prefetch(bSrc + k0 + 32, 0, 3);
        v8h h;
#pragma unroll
        for (int i = 0; i < 4; ++i) {
            h[i]     = (_Float16)(bValid ? f0[i] : 0.0f);
            h[i + 4] = (_Float16)(bValid ? f1[i] : 0.0f);
        }
        return h;
    };
    auto storeB = [&](int buf, v8h h) {
        *(v8h*)&lB[buf][ldsOff] = h;
    };

    // ---- prologue: stage tile 0 ----
    asyncA(0, 0);
    storeB(0, loadB(0));

    const int nk = K >> 5;   // K multiple of 32
    for (int i = 0; i < nk; ++i) {
        const int cur = i & 1;
        wait_async0();        // this wave's A-tile portion has landed in LDS
        __syncthreads();      // all waves' A async + B ds_stores visible

        const bool more = (i + 1) < nk;
        v8h hn = {};
        if (more) {
            asyncA(cur ^ 1, (i + 1) << 5);   // overlap next A with compute
            hn = loadB((i + 1) << 5);
        }

        // ---- B fragment (32x16): lanes 0-15 K0..7,16..23, lanes 16-31 K8..15,24..31 ----
        const _Float16* pb = &lB[cur][(wn * 16 + l16) * LDSH + hb * 8];
        v16h bf = cat8(*(const v8h*)pb, *(const v8h*)(pb + 16));

        // ---- A fragments + WMMA ----
        {
            const _Float16* pa = &lA[cur][(wm * 32 + l16) * LDSH + hb * 8];
            v16h af = cat8(*(const v8h*)pa, *(const v8h*)(pa + 16));
            acc0 = __builtin_amdgcn_wmma_f32_16x16x32_f16(
                false, af, false, bf, (short)0, acc0, false, false);
        }
        {
            const _Float16* pa = &lA[cur][(wm * 32 + 16 + l16) * LDSH + hb * 8];
            v16h af = cat8(*(const v8h*)pa, *(const v8h*)(pa + 16));
            acc1 = __builtin_amdgcn_wmma_f32_16x16x32_f16(
                false, af, false, bf, (short)0, acc1, false, false);
        }

        if (more) storeB(cur ^ 1, hn);       // other buffer: no barrier needed
    }

    // ---- epilogue ----
    const int col = nBase + wn * 16 + l16;
    const float bv = (col < N) ? bias[col] : 0.0f;
    const int rb = mBase + wm * 32 + hb * 8;

    if (RELU_F16OUT) {
        _Float16* C = (_Float16*)Cout;
#pragma unroll
        for (int v = 0; v < 8; ++v) {
            float va = fmaxf(acc0[v] + bv, 0.0f);
            float vb = fmaxf(acc1[v] + bv, 0.0f);
            if (col < N) {
                C[(size_t)(rb + v) * ldc + col]      = (_Float16)va;
                C[(size_t)(rb + 16 + v) * ldc + col] = (_Float16)vb;
            }
        }
    } else {
        float* C = (float*)Cout;
#pragma unroll
        for (int v = 0; v < 8; ++v) {
            if (col < N) {
                C[(size_t)(rb + v) * ldc + col]      = acc0[v] + bv;
                C[(size_t)(rb + 16 + v) * ldc + col] = acc1[v] + bv;
            }
        }
    }
}

// ---------------------------------------------------------------------------
// Kernel 6: softmax over cls logits + output assembly
// out = [rois 2560][cls_probs 512*81][bbox 512*324]
// grid = 64 blocks x 256 threads; one wave32 per roi row
// ---------------------------------------------------------------------------
__global__ void head_finalize(const float* __restrict__ rois,
                              const float* __restrict__ logits,
                              float* __restrict__ out) {
    const int tid = blockIdx.x * 256 + threadIdx.x;
    if (tid < N_ROIS * 5) out[tid] = rois[tid];   // rois_out passthrough

    const int wave = threadIdx.x >> 5;
    const int lane = threadIdx.x & 31;
    const int row = blockIdx.x * 8 + wave;        // 0..511
    const float* lg = logits + (size_t)row * NLOG;

    // --- softmax over 81 cls logits ---
    const float NEG = -3.0e38f;
    float v0 = (lane      < NCLS) ? lg[lane]      : NEG;
    float v1 = (lane + 32 < NCLS) ? lg[lane + 32] : NEG;
    float v2 = (lane + 64 < NCLS) ? lg[lane + 64] : NEG;
    float mx = fmaxf(v0, fmaxf(v1, v2));
#pragma unroll
    for (int off = 16; off >= 1; off >>= 1)
        mx = fmaxf(mx, __shfl_xor(mx, off, 32));

    float e0 = (lane      < NCLS) ? expf(v0 - mx) : 0.0f;
    float e1 = (lane + 32 < NCLS) ? expf(v1 - mx) : 0.0f;
    float e2 = (lane + 64 < NCLS) ? expf(v2 - mx) : 0.0f;
    float sm = e0 + e1 + e2;
#pragma unroll
    for (int off = 16; off >= 1; off >>= 1)
        sm += __shfl_xor(sm, off, 32);
    const float inv = 1.0f / sm;

    float* oc = out + N_ROIS * 5 + (size_t)row * NCLS;
    if (lane      < NCLS) oc[lane]      = e0 * inv;
    if (lane + 32 < NCLS) oc[lane + 32] = e1 * inv;
    if (lane + 64 < NCLS) oc[lane + 64] = e2 * inv;

    // --- bbox preds (bias already added in GEMM) ---
    float* ob = out + N_ROIS * 5 + N_ROIS * NCLS + (size_t)row * NREG;
    for (int e = lane; e < NREG; e += 32) ob[e] = lg[NCLS + e];
}

// ---------------------------------------------------------------------------
// Launch
// ---------------------------------------------------------------------------
extern "C" void kernel_launch(void* const* d_in, const int* in_sizes, int n_in,
                              void* d_out, int out_size, void* d_ws, size_t ws_size,
                              hipStream_t stream) {
    const float* p2    = (const float*)d_in[0];
    const float* p3    = (const float*)d_in[1];
    const float* p4    = (const float*)d_in[2];
    const float* p5    = (const float*)d_in[3];
    const float* rois  = (const float*)d_in[4];
    const float* w1    = (const float*)d_in[5];
    const float* b1    = (const float*)d_in[6];
    const float* w2    = (const float*)d_in[7];
    const float* b2    = (const float*)d_in[8];
    const float* wcls  = (const float*)d_in[9];
    const float* bcls  = (const float*)d_in[10];
    const float* wreg  = (const float*)d_in[11];
    const float* breg  = (const float*)d_in[12];

    char* ws = (char*)d_ws;
    _Float16* x0 = (_Float16*)(ws);                         // 512*12544*2 = 12,845,056
    _Float16* x1 = (_Float16*)(ws + 12845056);              // 512*1024*2  =  1,048,576
    _Float16* x2 = (_Float16*)(ws + 12845056 + 1048576);    //              1,048,576
    float* logits = (float*)(ws + 12845056 + 2097152);      // 512*405*4   =    829,440

    float* out = (float*)d_out;

    // 1) ROI align -> x0 [512, 12544] f16
    roi_align_kernel<<<N_ROIS, 256, 0, stream>>>(p2, p3, p4, p5, rois, x0);

    // 2) FC1: x1 = relu(x0 @ w1^T + b1)    M=512 N=1024 K=12544
    gemm_nt<true><<<dim3(16, 8), 256, 0, stream>>>(
        x0, w1, b1, (void*)x1, HID, DFEAT, DFEAT, DFEAT, HID);

    // 3) FC2: x2 = relu(x1 @ w2^T + b2)    M=512 N=1024 K=1024
    gemm_nt<true><<<dim3(16, 8), 256, 0, stream>>>(
        x1, w2, b2, (void*)x2, HID, HID, HID, HID, HID);

    // 4) cls head -> logits[:, 0:81]       M=512 N=81 K=1024
    gemm_nt<false><<<dim3(2, 8), 256, 0, stream>>>(
        x2, wcls, bcls, (void*)logits, NCLS, HID, HID, HID, NLOG);

    // 5) reg head -> logits[:, 81:405]     M=512 N=324 K=1024
    gemm_nt<false><<<dim3(6, 8), 256, 0, stream>>>(
        x2, wreg, breg, (void*)(logits + NCLS), NREG, HID, HID, HID, NLOG);

    // 6) softmax + assemble outputs
    head_finalize<<<64, 256, 0, stream>>>(rois, logits, out);
}
